// MultiHeadAttention_27608049778751
// MI455X (gfx1250) — matrix operand order repaired and checked by reading, not compiled, not run on hardware
//
#include <hip/hip_runtime.h>
#include <hip/hip_bf16.h>

// MI455X / gfx1250, wave32, WMMA bf16 path.
// MHA forward: QKV GEMMs -> causal flash attention -> output projection.
// GEMMs are double-buffered (global loads for tile k+1 overlap wmma on tile k).
// Pure-copy staging uses GLOBAL_LOAD_ASYNC_TO_LDS_B128 (ASYNCcnt-tracked).

typedef __attribute__((ext_vector_type(16))) __bf16 v16bf;
typedef __attribute__((ext_vector_type(8)))  __bf16 v8bf;
typedef __attribute__((ext_vector_type(4)))  __bf16 v4bf;
typedef __attribute__((ext_vector_type(8)))  float  v8f;
typedef __attribute__((ext_vector_type(4)))  int    v4i;

#define B_   4
#define S_   2048
#define D_   2048
#define H_   16
#define DH_  128
#define MTOT (B_ * S_)        // 8192 rows for the big GEMMs

#if defined(__AMDGCN__) && __has_builtin(__builtin_amdgcn_global_load_async_to_lds_b128)
#define USE_ASYNC_LDS 1
#else
#define USE_ASYNC_LDS 0
#endif

#if USE_ASYNC_LDS
typedef __attribute__((address_space(1))) v4i gas_v4i;
typedef __attribute__((address_space(3))) v4i las_v4i;
__device__ __forceinline__ void async_copy16(const void* g, void* l) {
  // 16 bytes per lane, memory -> LDS, tracked by ASYNCcnt.
  __builtin_amdgcn_global_load_async_to_lds_b128(
      (gas_v4i*)(v4i*)g, (las_v4i*)(v4i*)l, 0, 0);
}
__device__ __forceinline__ void async_wait0() {
#if __has_builtin(__builtin_amdgcn_s_wait_asynccnt)
  __builtin_amdgcn_s_wait_asynccnt(0);
#else
  asm volatile("s_wait_asynccnt 0x0" ::: "memory");
#endif
}
#endif

union Frag { v16bf v; v8bf h[2]; };

__device__ __forceinline__ v8f wmma_bf16(const v16bf& a, const v16bf& b, const v8f& c) {
  // D = A(16x32 bf16) * B(32x16 bf16) + C(16x16 f32)
  return __builtin_amdgcn_wmma_f32_16x16x32_bf16(
      /*neg_a=*/false, a, /*neg_b=*/false, b,
      /*c_mod=*/(short)0, c, /*reuse_a=*/false, /*reuse_b=*/false);
}

// ---------------------------------------------------------------------------
// fp32 -> bf16 weight conversion (one-time, bandwidth trivial vs GEMM work)
// ---------------------------------------------------------------------------
__global__ __launch_bounds__(256) void cvt_f32_bf16(const float* __restrict__ in,
                                                    __bf16* __restrict__ out, int n) {
  int i = (blockIdx.x * blockDim.x + threadIdx.x) * 4;
  if (i + 3 < n) {
    float4 v = *(const float4*)(in + i);
    v4bf o;
    o[0] = (__bf16)v.x; o[1] = (__bf16)v.y; o[2] = (__bf16)v.z; o[3] = (__bf16)v.w;
    *(v4bf*)(out + i) = o;
  }
}

// ---------------------------------------------------------------------------
// Tiled GEMM: C[M x N] = A[M x K] * W[K x N]
// 256 threads (8 waves), 128x128 tile, K-step 32, double-buffered LDS.
// Wave grid 2(M) x 4(N): each wave owns 8 wmma accumulators (64x32 slab).
// A_F32: A is fp32 (x), converted to bf16 while staging.
// OUT_MODE 0: bf16 output scattered into [b,h,s,dh] head layout, scaled.
// OUT_MODE 1: fp32 output, + bias, flat [M x N].
// ---------------------------------------------------------------------------
template <int A_F32, int OUT_MODE>
__global__ __launch_bounds__(256) void gemm128(const void* __restrict__ Aptr,
                                               const __bf16* __restrict__ Bw,
                                               void* __restrict__ Out,
                                               const float* __restrict__ bias,
                                               float scale) {
  __shared__ __bf16 ldsA[2][128][40];   // [buf][M][Kstep + pad]
  __shared__ __bf16 ldsBt[2][128][40];  // [buf][N][Kstep + pad] (transposed W)

  const int tid  = threadIdx.x;
  const int lane = tid & 31;
  const int wave = tid >> 5;
  const int ln   = lane & 15;
  const int half = lane >> 4;
  const int wm   = wave & 1;   // 64-row slab
  const int wn   = wave >> 1;  // 32-col slab
  const int bm   = blockIdx.x;
  const int bn   = blockIdx.y;
  const int K    = D_;
  const int NK   = K / 32;

  // staging indices
  const int r0f = tid >> 3, c0f = (tid & 7) * 4;   // fp32-A path: 4 rows x float4
  const int rb  = tid >> 1, cb  = (tid & 1) * 16;  // bf16-A path: 1 row x 2*v8bf
  const int k0  = tid >> 4, n0  = (tid & 15) * 8;  // W path: 2 k-rows x v8bf

  float4 af[4];
  v8bf   ab[2];
  v8bf   wv[2];

  v8f acc[4][2];
#pragma unroll
  for (int i = 0; i < 4; ++i)
#pragma unroll
    for (int j = 0; j < 2; ++j)
#pragma unroll
      for (int e = 0; e < 8; ++e) acc[i][j][e] = 0.f;

  auto loadA = [&](int kt) {
    if (A_F32) {
      const float* A = (const float*)Aptr;
#pragma unroll
      for (int i = 0; i < 4; ++i)
        af[i] = *(const float4*)(A + (size_t)(bm * 128 + r0f + i * 32) * K + kt * 32 + c0f);
    } else {
#if !USE_ASYNC_LDS
      const __bf16* A = (const __bf16*)Aptr;
#pragma unroll
      for (int i = 0; i < 2; ++i)
        ab[i] = *(const v8bf*)(A + (size_t)(bm * 128 + rb) * K + kt * 32 + cb + i * 8);
#endif
    }
  };
#if USE_ASYNC_LDS
  auto asyncA = [&](int kt, int buf) {
    if (!A_F32) {
      const __bf16* A = (const __bf16*)Aptr;
#pragma unroll
      for (int i = 0; i < 2; ++i)
        async_copy16(A + (size_t)(bm * 128 + rb) * K + kt * 32 + cb + i * 8,
                     &ldsA[buf][rb][cb + i * 8]);
    }
  };
#endif
  auto loadW = [&](int kt) {
#pragma unroll
    for (int i = 0; i < 2; ++i)
      wv[i] = *(const v8bf*)(Bw + (size_t)(kt * 32 + k0 + i * 16) * D_ + bn * 128 + n0);
  };
  auto storeLDS = [&](int buf) {
    if (A_F32) {
#pragma unroll
      for (int i = 0; i < 4; ++i) {
        const int r = r0f + i * 32;
        ldsA[buf][r][c0f + 0] = (__bf16)af[i].x;
        ldsA[buf][r][c0f + 1] = (__bf16)af[i].y;
        ldsA[buf][r][c0f + 2] = (__bf16)af[i].z;
        ldsA[buf][r][c0f + 3] = (__bf16)af[i].w;
      }
    } else {
#if !USE_ASYNC_LDS
#pragma unroll
      for (int i = 0; i < 2; ++i) *(v8bf*)&ldsA[buf][rb][cb + i * 8] = ab[i];
#endif
    }
#pragma unroll
    for (int i = 0; i < 2; ++i) {
      const int k = k0 + i * 16;
#pragma unroll
      for (int j = 0; j < 8; ++j) ldsBt[buf][n0 + j][k] = wv[i][j];
    }
  };

  // ---- prologue: stage tile 0 into buffer 0 ----
  loadA(0);
  loadW(0);
#if USE_ASYNC_LDS
  if (!A_F32) asyncA(0, 0);
#endif
  storeLDS(0);
#if USE_ASYNC_LDS
  if (!A_F32) async_wait0();
#endif
  __syncthreads();

  // ---- main pipelined loop: one barrier per K-step ----
  for (int kt = 0; kt < NK; ++kt) {
    const int cur = kt & 1;
    const bool more = (kt + 1 < NK);
    if (more) {
      loadA(kt + 1);
      loadW(kt + 1);
#if USE_ASYNC_LDS
      if (!A_F32) asyncA(kt + 1, cur ^ 1);
#endif
      if (A_F32)
        __builtin_prefetch((const float*)Aptr + (size_t)(bm * 128 + r0f) * K + (kt + 1) * 32 + c0f, 0, 1);
    }

    // compute tile kt from lds[cur] while tile kt+1 loads are in flight
    Frag bfr[2];
#pragma unroll
    for (int ni = 0; ni < 2; ++ni) {
      const int n = wn * 32 + ni * 16 + ln;
      bfr[ni].h[0] = *(const v8bf*)&ldsBt[cur][n][half * 8];
      bfr[ni].h[1] = *(const v8bf*)&ldsBt[cur][n][half * 8 + 16];
    }
#pragma unroll
    for (int mi = 0; mi < 4; ++mi) {
      Frag afr;
      const int r = wm * 64 + mi * 16 + ln;
      afr.h[0] = *(const v8bf*)&ldsA[cur][r][half * 8];
      afr.h[1] = *(const v8bf*)&ldsA[cur][r][half * 8 + 16];
#pragma unroll
      for (int ni = 0; ni < 2; ++ni)
        acc[mi][ni] = wmma_bf16(afr.v, bfr[ni].v, acc[mi][ni]);
    }

    if (more) storeLDS(cur ^ 1);
#if USE_ASYNC_LDS
    if (!A_F32 && more) async_wait0();
#endif
    __syncthreads();
  }

  // ---- epilogue: C layout VGPR j -> M = j + 8*half, N = ln ----
#pragma unroll
  for (int mi = 0; mi < 4; ++mi) {
#pragma unroll
    for (int ni = 0; ni < 2; ++ni) {
#pragma unroll
      for (int j = 0; j < 8; ++j) {
        const int row = bm * 128 + wm * 64 + mi * 16 + j + 8 * half;
        const int col = bn * 128 + wn * 32 + ni * 16 + ln;
        const float v = acc[mi][ni][j];
        if (OUT_MODE == 0) {
          const int b = row >> 11, s = row & (S_ - 1);
          const int h = col >> 7, dh = col & (DH_ - 1);
          ((__bf16*)Out)[(((size_t)(b * H_ + h)) * S_ + s) * DH_ + dh] = (__bf16)(v * scale);
        } else {
          ((float*)Out)[(size_t)row * D_ + col] = v + bias[col];
        }
      }
    }
  }
}

// ---------------------------------------------------------------------------
// Flash attention: one block per (b, h, 128-row q-tile). 8 waves, each owns
// 16 q-rows. Online softmax fully in registers via 16-lane __shfl_xor trees.
// ---------------------------------------------------------------------------
__global__ __launch_bounds__(256) void attn_kernel(const __bf16* __restrict__ Qg,
                                                   const __bf16* __restrict__ Kg,
                                                   const __bf16* __restrict__ Vg,
                                                   __bf16* __restrict__ ctx) {
  extern __shared__ __align__(16) char smem_raw[];
  __bf16* ldsK  = (__bf16*)smem_raw;        // [128][136]  K tile, row-major (= B^T for Q*K^T)
  __bf16* ldsVt = ldsK + 128 * 136;         // [128][136]  V^T tile: [dh][kv]
  __bf16* ldsP  = ldsVt + 128 * 136;        // per-wave [16][136] P tiles

  const int tid  = threadIdx.x;
  const int lane = tid & 31;
  const int wave = tid >> 5;
  const int ln   = lane & 15;
  const int half = lane >> 4;

  int bid = blockIdx.x;
  const int qtile = bid & 15;  bid >>= 4;
  const int h     = bid & 15;  bid >>= 4;
  const int b     = bid;

  const size_t bh_off = ((size_t)(b * H_ + h)) * S_ * DH_;
  const int qbase = qtile * 128 + wave * 16;

  // Q fragments stay in registers for the whole kernel (4 K-steps over DH=128)
  Frag qf[4];
  {
    const __bf16* qrow = Qg + bh_off + (size_t)(qbase + ln) * DH_;
#pragma unroll
    for (int ks = 0; ks < 4; ++ks) {
      qf[ks].h[0] = *(const v8bf*)(qrow + ks * 32 + half * 8);
      qf[ks].h[1] = *(const v8bf*)(qrow + ks * 32 + 16 + half * 8);
    }
  }

  v8f cacc[8];
#pragma unroll
  for (int d = 0; d < 8; ++d)
#pragma unroll
    for (int e = 0; e < 8; ++e) cacc[d][e] = 0.f;
  float mrun[8], lrun[8];
#pragma unroll
  for (int j = 0; j < 8; ++j) { mrun[j] = -__builtin_inff(); lrun[j] = 0.f; }

  for (int kt = 0; kt <= qtile; ++kt) {   // causal: lower-triangular tiles only
    // ---- stage K (async copy) and V (transposed, manual) ----
    {
      const int r  = tid >> 1;
      const int cb = (tid & 1) * 64;
      const __bf16* krow = Kg + bh_off + (size_t)(kt * 128 + r) * DH_ + cb;
      const __bf16* vrow = Vg + bh_off + (size_t)(kt * 128 + r) * DH_ + cb;
#if USE_ASYNC_LDS
#pragma unroll
      for (int i = 0; i < 8; ++i)
        async_copy16(krow + i * 8, &ldsK[r * 136 + cb + i * 8]);
#else
#pragma unroll
      for (int i = 0; i < 8; ++i)
        *(v8bf*)&ldsK[r * 136 + cb + i * 8] = *(const v8bf*)(krow + i * 8);
#endif
#pragma unroll
      for (int i = 0; i < 8; ++i) {
        v8bf vv8 = *(const v8bf*)(vrow + i * 8);
#pragma unroll
        for (int j = 0; j < 8; ++j) ldsVt[(size_t)(cb + i * 8 + j) * 136 + r] = vv8[j];
      }
      if (kt < qtile) {
        __builtin_prefetch(Kg + bh_off + (size_t)((kt + 1) * 128 + r) * DH_ + cb, 0, 1);
        __builtin_prefetch(Vg + bh_off + (size_t)((kt + 1) * 128 + r) * DH_ + cb, 0, 1);
      }
#if USE_ASYNC_LDS
      async_wait0();
#endif
    }
    __syncthreads();

    // ---- S = Q * K^T : 8 col-tiles x 4 K-steps = 32 wmma ----
    v8f sacc[8];
#pragma unroll
    for (int t = 0; t < 8; ++t) {
#pragma unroll
      for (int e = 0; e < 8; ++e) sacc[t][e] = 0.f;
#pragma unroll
      for (int ks = 0; ks < 4; ++ks) {
        Frag bf;
        const __bf16* p = &ldsK[(size_t)(t * 16 + ln) * 136 + ks * 32 + half * 8];
        bf.h[0] = *(const v8bf*)p;
        bf.h[1] = *(const v8bf*)(p + 16);
        sacc[t] = wmma_bf16(qf[ks].v, bf.v, sacc[t]);
      }
    }

    // ---- causal mask on the diagonal tile ----
    if (kt == qtile) {
#pragma unroll
      for (int t = 0; t < 8; ++t)
#pragma unroll
        for (int j = 0; j < 8; ++j) {
          const int colr = t * 16 + ln;
          const int rowr = wave * 16 + j + 8 * half;
          if (colr > rowr) sacc[t][j] = -__builtin_inff();
        }
    }

    // ---- online softmax, register-resident (16-lane shfl_xor trees) ----
    float alpha[8];
#pragma unroll
    for (int j = 0; j < 8; ++j) {
      float m = sacc[0][j];
#pragma unroll
      for (int t = 1; t < 8; ++t) m = fmaxf(m, sacc[t][j]);
#pragma unroll
      for (int off = 1; off < 16; off <<= 1) m = fmaxf(m, __shfl_xor(m, off, 32));
      const float mnew = fmaxf(mrun[j], m);
      const float a = __expf(mrun[j] - mnew);
      alpha[j] = a;
      mrun[j] = mnew;
      float rs = 0.f;
#pragma unroll
      for (int t = 0; t < 8; ++t) {
        const float p = __expf(sacc[t][j] - mnew);
        sacc[t][j] = p;
        rs += p;
      }
#pragma unroll
      for (int off = 1; off < 16; off <<= 1) rs += __shfl_xor(rs, off, 32);
      lrun[j] = lrun[j] * a + rs;
    }
#pragma unroll
    for (int d = 0; d < 8; ++d)
#pragma unroll
      for (int j = 0; j < 8; ++j) cacc[d][j] *= alpha[j];

    // ---- reshape P through LDS into the A-fragment layout ----
    __bf16* pw = ldsP + (size_t)wave * 16 * 136;
#pragma unroll
    for (int t = 0; t < 8; ++t)
#pragma unroll
      for (int j = 0; j < 8; ++j)
        pw[(size_t)(j + 8 * half) * 136 + t * 16 + ln] = (__bf16)sacc[t][j];
    __syncthreads();

    // ---- ctx += P * V : 8 dh-tiles x 4 K-steps = 32 wmma ----
    Frag pf[4];
#pragma unroll
    for (int ks = 0; ks < 4; ++ks) {
      const __bf16* pp = pw + (size_t)ln * 136 + ks * 32 + half * 8;
      pf[ks].h[0] = *(const v8bf*)pp;
      pf[ks].h[1] = *(const v8bf*)(pp + 16);
    }
#pragma unroll
    for (int d = 0; d < 8; ++d) {
#pragma unroll
      for (int ks = 0; ks < 4; ++ks) {
        Frag bf;
        const __bf16* vp = &ldsVt[(size_t)(d * 16 + ln) * 136 + ks * 32 + half * 8];
        bf.h[0] = *(const v8bf*)vp;
        bf.h[1] = *(const v8bf*)(vp + 16);
        cacc[d] = wmma_bf16(pf[ks].v, bf.v, cacc[d]);
      }
    }
    __syncthreads();
  }

  // ---- epilogue: ctx[b][s][h*128 + dh] = acc / l ----
#pragma unroll
  for (int d = 0; d < 8; ++d) {
#pragma unroll
    for (int j = 0; j < 8; ++j) {
      const int srow = qbase + j + 8 * half;
      const float v = cacc[d][j] / lrun[j];
      ctx[((size_t)(b * S_ + srow)) * D_ + h * DH_ + d * 16 + ln] = (__bf16)v;
    }
  }
}

// ---------------------------------------------------------------------------
// Host-side orchestration
// ---------------------------------------------------------------------------
extern "C" void kernel_launch(void* const* d_in, const int* in_sizes, int n_in,
                              void* d_out, int out_size, void* d_ws, size_t ws_size,
                              hipStream_t stream) {
  (void)in_sizes; (void)n_in; (void)out_size; (void)ws_size;

  const float* x  = (const float*)d_in[0];
  const float* Wq = (const float*)d_in[1];
  const float* Wk = (const float*)d_in[2];
  const float* Wv = (const float*)d_in[3];
  const float* Wo = (const float*)d_in[4];
  const float* bo = (const float*)d_in[5];

  char* ws = (char*)d_ws;
  size_t off = 0;
  auto carve = [&](size_t bytes) -> void* {
    void* p = ws + off;
    off += (bytes + 255) & ~(size_t)255;
    return p;
  };
  const size_t wElems = (size_t)D_ * D_;
  const size_t qkvElems = (size_t)B_ * H_ * S_ * DH_;
  __bf16* Wq_b = (__bf16*)carve(wElems * 2);
  __bf16* Wk_b = (__bf16*)carve(wElems * 2);
  __bf16* Wv_b = (__bf16*)carve(wElems * 2);
  __bf16* Wo_b = (__bf16*)carve(wElems * 2);
  __bf16* Qb   = (__bf16*)carve(qkvElems * 2);
  __bf16* Kb   = (__bf16*)carve(qkvElems * 2);
  __bf16* Vb   = (__bf16*)carve(qkvElems * 2);
  __bf16* ctx  = (__bf16*)carve((size_t)B_ * S_ * D_ * 2);

  // 1) weights fp32 -> bf16
  {
    const int n = (int)wElems;
    dim3 grd(n / (256 * 4)), blk(256);
    cvt_f32_bf16<<<grd, blk, 0, stream>>>(Wq, Wq_b, n);
    cvt_f32_bf16<<<grd, blk, 0, stream>>>(Wk, Wk_b, n);
    cvt_f32_bf16<<<grd, blk, 0, stream>>>(Wv, Wv_b, n);
    cvt_f32_bf16<<<grd, blk, 0, stream>>>(Wo, Wo_b, n);
  }

  // 2) QKV projections (1/sqrt(DH) folded into Q)
  {
    dim3 grd(MTOT / 128, D_ / 128), blk(256);
    const float qscale = 0.08838834764831845f;  // 1/sqrt(128)
    gemm128<1, 0><<<grd, blk, 0, stream>>>(x, Wq_b, Qb, nullptr, qscale);
    gemm128<1, 0><<<grd, blk, 0, stream>>>(x, Wk_b, Kb, nullptr, 1.0f);
    gemm128<1, 0><<<grd, blk, 0, stream>>>(x, Wv_b, Vb, nullptr, 1.0f);
  }

  // 3) causal flash attention
  {
    const int smem = (128 * 136 + 128 * 136 + 8 * 16 * 136) * 2;  // 104448 B
    hipError_t err = hipFuncSetAttribute(reinterpret_cast<const void*>(&attn_kernel),
                                         hipFuncAttributeMaxDynamicSharedMemorySize, smem);
    (void)err;
    attn_kernel<<<dim3(B_ * H_ * (S_ / 128)), dim3(256), smem, stream>>>(Qb, Kb, Vb, ctx);
  }

  // 4) output projection + bias -> fp32 d_out
  {
    dim3 grd(MTOT / 128, D_ / 128), blk(256);
    gemm128<0, 1><<<grd, blk, 0, stream>>>(ctx, Wo_b, d_out, bo, 1.0f);
  }
}
